// MultiHeadAttention_23450521436337
// MI455X (gfx1250) — matrix operand order repaired and checked by reading, not compiled, not run on hardware
//
#include <hip/hip_runtime.h>

// ---------------------------------------------------------------------------
// MHA block for MI455X (gfx1250): bf16 WMMA GEMMs + flash attention with
// TDM (tensor_load_to_lds) K/V tile staging.
// B=2, S=2048, D=1024, H=16, DK=64. Compute-bound (~103 GFLOP vs ~150MB).
// ---------------------------------------------------------------------------

#define BB   2
#define SS   2048
#define DD   1024
#define HH   16
#define DKK  64
#define BS   (BB*SS)        // 4096 rows

typedef __attribute__((ext_vector_type(16))) __bf16        v16bf;
typedef __attribute__((ext_vector_type(8)))  float         v8f;
typedef __attribute__((ext_vector_type(4)))  unsigned int  u32x4;
typedef __attribute__((ext_vector_type(8)))  int           i32x8;
typedef __attribute__((ext_vector_type(4)))  int           i32x4;
typedef __attribute__((ext_vector_type(4)))  float         f32x4;

#ifdef __has_builtin
#  if __has_builtin(__builtin_amdgcn_tensor_load_to_lds)
#    define HAVE_TDM 1
#  endif
#endif
#ifndef HAVE_TDM
#  define HAVE_TDM 0
#endif

union FragU {
    v16bf          v;      // WMMA operand (8 VGPRs)
    unsigned short us[16];
    u32x4          q[2];   // two 16B chunks
};

__device__ __forceinline__ unsigned short f2bf(float f) {
    __bf16 h = (__bf16)f;                     // native v_cvt path on gfx1250
    return __builtin_bit_cast(unsigned short, h);
}

__device__ __forceinline__ void cvt8(const float* __restrict__ p, unsigned short* d) {
    f32x4 a = *(const f32x4*)p;
    f32x4 b = *(const f32x4*)(p + 4);
    d[0]=f2bf(a.x); d[1]=f2bf(a.y); d[2]=f2bf(a.z); d[3]=f2bf(a.w);
    d[4]=f2bf(b.x); d[5]=f2bf(b.y); d[6]=f2bf(b.z); d[7]=f2bf(b.w);
}

__device__ __forceinline__ v8f wmma_bf16(v16bf a, v16bf b, v8f c) {
    // v_wmma_f32_16x16x32_bf16  (neg_a, A, neg_b, B, c_mod, C, reuse_a, reuse_b)
    return __builtin_amdgcn_wmma_f32_16x16x32_bf16(false, a, false, b, (short)0, c,
                                                   false, false);
}

#if HAVE_TDM
// 2D TDM load: tile (tile_d1 rows x tile_d0 elems) of 2-byte elements from
// global (row stride stride_d0 elems) into contiguous LDS at lds_off.
// D# per CDNA5 ISA ch.8.3/8.4: group0 = {flags, lds_addr, gaddr_lo, gaddr_hi|type},
// group1 = {data_size, tensor dims, tile dims, dim0 stride}.
__device__ __forceinline__ void tdm_load_2d(unsigned lds_off, const void* gaddr,
                                            unsigned tile_d0, unsigned tile_d1,
                                            unsigned stride_d0)
{
    unsigned long long ga = (unsigned long long)(uintptr_t)gaddr;
    u32x4 g0;
    g0.x = 1u;                                            // count=1 (valid), user mode
    g0.y = lds_off;                                       // lds_addr [63:32]
    g0.z = (unsigned)(ga & 0xFFFFFFFFu);                  // global_addr[31:0]
    g0.w = (unsigned)((ga >> 32) & 0x01FFFFFFu) | (2u << 30); // addr[56:32] | type=2
    i32x8 g1;
    g1[0] = 0x00010000;                                   // data_size=1 -> 2 bytes
    g1[1] = (int)(tile_d0 << 16);                         // tensor_dim0 = tile_d0
    g1[2] = (int)((tile_d0 >> 16) | (tile_d1 << 16));     // tensor_dim1 = tile_d1
    g1[3] = (int)((tile_d1 >> 16) | (tile_d0 << 16));     // tile_dim0
    g1[4] = (int)(tile_d1 & 0xFFFFu);                     // tile_dim1 (tile_dim2=0)
    g1[5] = (int)stride_d0;                               // tensor_dim0_stride lo32
    g1[6] = 0;
    g1[7] = 0;
    i32x4 z4 = {0, 0, 0, 0};
#if __clang_major__ >= 23
    i32x8 z8 = {0, 0, 0, 0, 0, 0, 0, 0};
    __builtin_amdgcn_tensor_load_to_lds(g0, g1, z4, z4, z8, 0);
#else
    __builtin_amdgcn_tensor_load_to_lds(g0, g1, z4, z4, 0);
#endif
}
#endif

// ---------------------------------------------------------------------------
// Kernel 1: fused Q/K/V projection.  Y = X @ W^T + b, emitted in bf16.
//   blockIdx.z: 0=Q -> [B,H,S,DK], 1=K -> [B,H,S,DK], 2=V -> [B,H,DK,S]
// ---------------------------------------------------------------------------
__global__ __launch_bounds__(128)
void qkv_proj_kernel(const float* __restrict__ q, const float* __restrict__ k,
                     const float* __restrict__ v,
                     const float* __restrict__ Wq, const float* __restrict__ bq,
                     const float* __restrict__ Wk, const float* __restrict__ bk,
                     const float* __restrict__ Wv, const float* __restrict__ bv,
                     unsigned short* __restrict__ Qh, unsigned short* __restrict__ Kh,
                     unsigned short* __restrict__ Vt)
{
    const int which = blockIdx.z;
    const float* X; const float* W; const float* bias; unsigned short* out;
    if (which == 0)      { X = q; W = Wq; bias = bq; out = Qh; }
    else if (which == 1) { X = k; W = Wk; bias = bk; out = Kh; }
    else                 { X = v; W = Wv; bias = bv; out = Vt; }

    __shared__ unsigned short sW[64][32];   // [n_local][k_local], bf16

    const int tid   = threadIdx.x;
    const int wave  = tid >> 5;
    const int lane  = tid & 31;
    const int lhalf = lane >> 4;
    const int l16   = lane & 15;
    const int m0    = blockIdx.y * 64 + wave * 16;
    const int n0    = blockIdx.x * 64;

    v8f acc[4] = {};

    for (int k0 = 0; k0 < DD; k0 += 32) {
        // --- cooperative W tile load: W[n0+n][k0..k0+31] -> sW (bf16) ---
        {
            const int n = tid >> 1, half = tid & 1;
            const float* wr = W + (size_t)(n0 + n) * DD + k0 + half * 16;
            unsigned short tmp[16];
            cvt8(wr, tmp); cvt8(wr + 8, tmp + 8);
            *(u32x4*)&sW[n][half * 16]     = *(const u32x4*)&tmp[0];
            *(u32x4*)&sW[n][half * 16 + 8] = *(const u32x4*)&tmp[8];
            if (k0 + 32 < DD) __builtin_prefetch(wr + 32, 0, 1);  // global_prefetch_b8
        }
        __syncthreads();

        // --- A fragment: X rows (f32 -> bf16), one contiguous 16-run per lane ---
        FragU a;
        const float* xr = X + (size_t)(m0 + l16) * DD + k0 + lhalf * 16;
        cvt8(xr,      a.us);
        cvt8(xr + 8,  a.us + 8);

        #pragma unroll
        for (int j = 0; j < 4; ++j) {
            FragU bF;
            bF.q[0] = *(const u32x4*)&sW[j * 16 + l16][lhalf * 16];
            bF.q[1] = *(const u32x4*)&sW[j * 16 + l16][lhalf * 16 + 8];
            acc[j] = wmma_bf16(a.v, bF.v, acc[j]);
        }
        __syncthreads();
    }

    // --- epilogue: bias, layout shuffle, bf16 store ---
    #pragma unroll
    for (int j = 0; j < 4; ++j) {
        const int n  = n0 + j * 16 + l16;
        const float bia = bias[n];
        const int h  = n >> 6, dk = n & 63;
        #pragma unroll
        for (int r = 0; r < 8; ++r) {
            const int m  = m0 + r + lhalf * 8;
            const int bb = m >> 11, s = m & (SS - 1);
            const float val = acc[j][r] + bia;
            size_t off;
            if (which == 2) off = (((size_t)bb * HH + h) * DKK + dk) * SS + s; // [B,H,DK,S]
            else            off = (((size_t)bb * HH + h) * SS + s) * DKK + dk; // [B,H,S,DK]
            out[off] = f2bf(val);
        }
    }
}

// ---------------------------------------------------------------------------
// Kernel 2: flash attention. grid=(S/64, H, B), block=128 (4 waves).
// K/V 64x64 tiles staged in LDS once per block via TDM; each wave owns
// 16 q-rows and streams 64-key tiles with online softmax.
// ---------------------------------------------------------------------------
__global__ __launch_bounds__(128)
void attn_kernel(const unsigned short* __restrict__ Qh,
                 const unsigned short* __restrict__ Kh,
                 const unsigned short* __restrict__ Vt,
                 const int* __restrict__ mask,
                 unsigned short* __restrict__ ctx)
{
    const int b = blockIdx.z, h = blockIdx.y;
    const int tid   = threadIdx.x;
    const int wave  = tid >> 5;
    const int lane  = tid & 31;
    const int lhalf = lane >> 4;
    const int l16   = lane & 15;
    const int q0    = blockIdx.x * 64 + wave * 16;

    __shared__ unsigned short sK[64][64];      // [key][dk]  8 KB
    __shared__ unsigned short sV[64][64];      // [dk][key]  8 KB
    __shared__ unsigned short sP[4][16][64];   // per-wave P tile, 8 KB

    const unsigned short* Qbh = Qh + (((size_t)b * HH + h) * SS) * DKK;
    const unsigned short* Kbh = Kh + (((size_t)b * HH + h) * SS) * DKK;
    const unsigned short* Vbh = Vt + (((size_t)b * HH + h) * DKK) * SS;
    const int* maskb = mask + (size_t)b * SS * SS;

    // Q fragments for both 32-wide K-dim chunks (held for the whole pass)
    FragU qA[2];
    #pragma unroll
    for (int c = 0; c < 2; ++c) {
        const unsigned short* qr = Qbh + (size_t)(q0 + l16) * DKK + c * 32 + lhalf * 16;
        qA[c].q[0] = *(const u32x4*)qr;
        qA[c].q[1] = *(const u32x4*)(qr + 8);
    }

    v8f oacc[4] = {};
    float rowM[8], rowL[8];
    #pragma unroll
    for (int r = 0; r < 8; ++r) { rowM[r] = -3.0e38f; rowL[r] = 0.f; }

    for (int kt = 0; kt < SS / 64; ++kt) {
        const int key0 = kt * 64;

        // ---- stage K/V tiles into LDS (TDM, one DMA pair per block) ----
        __syncthreads();                        // previous tile fully consumed
#if HAVE_TDM
        if (wave == 0) {
            tdm_load_2d((unsigned)(uintptr_t)&sK[0][0],
                        Kbh + (size_t)key0 * DKK, DKK, 64, DKK);
            tdm_load_2d((unsigned)(uintptr_t)&sV[0][0],
                        Vbh + key0, 64, DKK, SS);
            __builtin_amdgcn_s_wait_tensorcnt(0);
        }
#else
        {   // fallback: cooperative copy, 2 threads per 128B row
            const int row = tid >> 1, half = tid & 1;
            const unsigned short* krow = Kbh + (size_t)(key0 + row) * DKK + half * 32;
            const unsigned short* vrow = Vbh + (size_t)row * SS + key0 + half * 32;
            #pragma unroll
            for (int t = 0; t < 4; ++t) {
                *(u32x4*)&sK[row][half * 32 + t * 8] = *(const u32x4*)(krow + t * 8);
                *(u32x4*)&sV[row][half * 32 + t * 8] = *(const u32x4*)(vrow + t * 8);
            }
        }
#endif
        __syncthreads();

        // ---- S = Q K^T (16x64 tile, 8 WMMAs), K frags from LDS ----
        v8f sacc[4] = {};
        #pragma unroll
        for (int c = 0; c < 2; ++c) {
            #pragma unroll
            for (int j = 0; j < 4; ++j) {
                FragU kB;
                kB.q[0] = *(const u32x4*)&sK[j * 16 + l16][c * 32 + lhalf * 16];
                kB.q[1] = *(const u32x4*)&sK[j * 16 + l16][c * 32 + lhalf * 16 + 8];
                sacc[j] = wmma_bf16(qA[c].v, kB.v, sacc[j]);
            }
        }

        // ---- scale, mask-fill, online softmax ----
        float p[4][8];
        float tmax[8];
        #pragma unroll
        for (int r = 0; r < 8; ++r) tmax[r] = -3.0e38f;
        #pragma unroll
        for (int j = 0; j < 4; ++j) {
            const int kn = key0 + j * 16 + l16;
            #pragma unroll
            for (int r = 0; r < 8; ++r) {
                const int qm = q0 + r + lhalf * 8;
                float sc = sacc[j][r] * 0.125f;                 // 1/sqrt(64)
                if (maskb[(size_t)qm * SS + kn] == 0) sc = -1.0e9f;
                p[j][r] = sc;
                tmax[r] = fmaxf(tmax[r], sc);
            }
        }
        // row-max across the 16 lanes of each half-group (C-layout: n = lane&15)
        #pragma unroll
        for (int d = 1; d < 16; d <<= 1)
            #pragma unroll
            for (int r = 0; r < 8; ++r)
                tmax[r] = fmaxf(tmax[r], __shfl_xor(tmax[r], d, 32));

        float newM[8], corr[8], tsum[8];
        #pragma unroll
        for (int r = 0; r < 8; ++r) {
            newM[r] = fmaxf(rowM[r], tmax[r]);
            corr[r] = __expf(rowM[r] - newM[r]);
            rowM[r] = newM[r];
            tsum[r] = 0.f;
        }
        #pragma unroll
        for (int j = 0; j < 4; ++j)
            #pragma unroll
            for (int r = 0; r < 8; ++r) {
                float e = __expf(p[j][r] - newM[r]);
                p[j][r] = e;
                tsum[r] += e;
            }
        #pragma unroll
        for (int d = 1; d < 16; d <<= 1)
            #pragma unroll
            for (int r = 0; r < 8; ++r)
                tsum[r] += __shfl_xor(tsum[r], d, 32);
        #pragma unroll
        for (int r = 0; r < 8; ++r) rowL[r] = rowL[r] * corr[r] + tsum[r];
        #pragma unroll
        for (int j = 0; j < 4; ++j)
            #pragma unroll
            for (int r = 0; r < 8; ++r)
                oacc[j][r] *= corr[r];

        // ---- P tile to LDS (C-layout -> A-layout shuffle), per-wave region ----
        #pragma unroll
        for (int j = 0; j < 4; ++j)
            #pragma unroll
            for (int r = 0; r < 8; ++r)
                sP[wave][r + lhalf * 8][j * 16 + l16] = f2bf(p[j][r]);
        asm volatile("s_wait_dscnt 0" ::: "memory");   // same-wave LDS RAW fence

        // ---- O += P V (V frags from LDS, 8 WMMAs) ----
        #pragma unroll
        for (int c = 0; c < 2; ++c) {
            FragU pA;
            const unsigned short* pr = &sP[wave][l16][c * 32 + lhalf * 16];
            pA.q[0] = *(const u32x4*)pr;
            pA.q[1] = *(const u32x4*)(pr + 8);
            #pragma unroll
            for (int j = 0; j < 4; ++j) {
                FragU vB;
                vB.q[0] = *(const u32x4*)&sV[j * 16 + l16][c * 32 + lhalf * 16];
                vB.q[1] = *(const u32x4*)&sV[j * 16 + l16][c * 32 + lhalf * 16 + 8];
                oacc[j] = wmma_bf16(pA.v, vB.v, oacc[j]);
            }
        }
    }

    // ---- epilogue: normalize, write ctx in concat layout [B,S,D] (bf16) ----
    #pragma unroll
    for (int j = 0; j < 4; ++j) {
        const int col = h * DKK + j * 16 + l16;
        #pragma unroll
        for (int r = 0; r < 8; ++r) {
            const int qm = q0 + r + lhalf * 8;
            const float val = oacc[j][r] / rowL[r];
            ctx[((size_t)b * SS + qm) * DD + col] = f2bf(val);
        }
    }
}

// ---------------------------------------------------------------------------
// Kernel 3: output projection + residual.  X = ctx @ Wo^T + bo + q
// ---------------------------------------------------------------------------
__global__ __launch_bounds__(128)
void out_proj_kernel(const unsigned short* __restrict__ ctx,
                     const float* __restrict__ Wo, const float* __restrict__ bo,
                     const float* __restrict__ qres, float* __restrict__ X)
{
    __shared__ unsigned short sW[64][32];

    const int tid   = threadIdx.x;
    const int wave  = tid >> 5;
    const int lane  = tid & 31;
    const int lhalf = lane >> 4;
    const int l16   = lane & 15;
    const int m0    = blockIdx.y * 64 + wave * 16;
    const int n0    = blockIdx.x * 64;

    v8f acc[4] = {};

    for (int k0 = 0; k0 < DD; k0 += 32) {
        {
            const int n = tid >> 1, half = tid & 1;
            const float* wr = Wo + (size_t)(n0 + n) * DD + k0 + half * 16;
            unsigned short tmp[16];
            cvt8(wr, tmp); cvt8(wr + 8, tmp + 8);
            *(u32x4*)&sW[n][half * 16]     = *(const u32x4*)&tmp[0];
            *(u32x4*)&sW[n][half * 16 + 8] = *(const u32x4*)&tmp[8];
            if (k0 + 32 < DD) __builtin_prefetch(wr + 32, 0, 1);
        }
        __syncthreads();

        FragU a;
        const unsigned short* ar = ctx + (size_t)(m0 + l16) * DD + k0 + lhalf * 16;
        a.q[0] = *(const u32x4*)ar;
        a.q[1] = *(const u32x4*)(ar + 8);

        #pragma unroll
        for (int j = 0; j < 4; ++j) {
            FragU bF;
            bF.q[0] = *(const u32x4*)&sW[j * 16 + l16][lhalf * 16];
            bF.q[1] = *(const u32x4*)&sW[j * 16 + l16][lhalf * 16 + 8];
            acc[j] = wmma_bf16(a.v, bF.v, acc[j]);
        }
        __syncthreads();
    }

    #pragma unroll
    for (int j = 0; j < 4; ++j) {
        const int n = n0 + j * 16 + l16;
        const float bia = bo[n];
        #pragma unroll
        for (int r = 0; r < 8; ++r) {
            const int m = m0 + r + lhalf * 8;
            X[(size_t)m * DD + n] = acc[j][r] + bia + qres[(size_t)m * DD + n];
        }
    }
}

// ---------------------------------------------------------------------------
// Kernel 4: LayerNorm over D=1024 per row. One 256-thread block per row.
// ---------------------------------------------------------------------------
__global__ __launch_bounds__(256)
void layernorm_kernel(const float* __restrict__ X,
                      const float* __restrict__ gamma, const float* __restrict__ beta,
                      float* __restrict__ out)
{
    const int row = blockIdx.x;
    const int tid = threadIdx.x;
    const float* xr = X + (size_t)row * DD;

    f32x4 v = *(const f32x4*)(xr + tid * 4);
    float s  = v.x + v.y + v.z + v.w;
    float ss = v.x * v.x + v.y * v.y + v.z * v.z + v.w * v.w;

    #pragma unroll
    for (int d = 1; d < 32; d <<= 1) {
        s  += __shfl_xor(s,  d, 32);
        ss += __shfl_xor(ss, d, 32);
    }

    __shared__ float aS[8], aSS[8];
    const int wave = tid >> 5, lane = tid & 31;
    if (lane == 0) { aS[wave] = s; aSS[wave] = ss; }
    __syncthreads();
    if (wave == 0) {
        float s2  = (lane < 8) ? aS[lane]  : 0.f;
        float ss2 = (lane < 8) ? aSS[lane] : 0.f;
        #pragma unroll
        for (int d = 1; d < 8; d <<= 1) {
            s2  += __shfl_xor(s2,  d, 32);
            ss2 += __shfl_xor(ss2, d, 32);
        }
        if (lane == 0) { aS[0] = s2; aSS[0] = ss2; }
    }
    __syncthreads();

    const float mu   = aS[0] * (1.f / DD);
    const float var  = aSS[0] * (1.f / DD) - mu * mu;
    const float rinv = rsqrtf(var + 1e-5f);

    const int c = tid * 4;
    f32x4 g  = *(const f32x4*)(gamma + c);
    f32x4 be = *(const f32x4*)(beta + c);
    f32x4 o;
    o.x = (v.x - mu) * rinv * g.x + be.x;
    o.y = (v.y - mu) * rinv * g.y + be.y;
    o.z = (v.z - mu) * rinv * g.z + be.z;
    o.w = (v.w - mu) * rinv * g.w + be.w;
    *(f32x4*)(out + (size_t)row * DD + c) = o;
}

// ---------------------------------------------------------------------------
extern "C" void kernel_launch(void* const* d_in, const int* in_sizes, int n_in,
                              void* d_out, int out_size, void* d_ws, size_t ws_size,
                              hipStream_t stream)
{
    const float* q     = (const float*)d_in[0];
    const float* k     = (const float*)d_in[1];
    const float* v     = (const float*)d_in[2];
    const int*   mask  = (const int*)  d_in[3];
    const float* Wq    = (const float*)d_in[4];
    const float* bq    = (const float*)d_in[5];
    const float* Wk    = (const float*)d_in[6];
    const float* bk    = (const float*)d_in[7];
    const float* Wv    = (const float*)d_in[8];
    const float* bv    = (const float*)d_in[9];
    const float* Wo    = (const float*)d_in[10];
    const float* bo    = (const float*)d_in[11];
    const float* gamma = (const float*)d_in[12];
    const float* beta  = (const float*)d_in[13];

    char* ws = (char*)d_ws;
    const size_t MB8 = 8ull << 20;
    unsigned short* Qh  = (unsigned short*)(ws);            //  8 MB  [B,H,S,DK]  bf16
    unsigned short* Kh  = (unsigned short*)(ws + 1 * MB8);  //  8 MB  [B,H,S,DK]  bf16
    unsigned short* Vt  = (unsigned short*)(ws + 2 * MB8);  //  8 MB  [B,H,DK,S]  bf16
    unsigned short* ctx = (unsigned short*)(ws + 3 * MB8);  //  8 MB  [B,S,D]     bf16
    float*          Xb  = (float*)        (ws + 4 * MB8);   // 16 MB  [B,S,D]     f32

    qkv_proj_kernel<<<dim3(DD / 64, BS / 64, 3), 128, 0, stream>>>(
        q, k, v, Wq, bq, Wk, bk, Wv, bv, Qh, Kh, Vt);

    attn_kernel<<<dim3(SS / 64, HH, BB), 128, 0, stream>>>(Qh, Kh, Vt, mask, ctx);

    out_proj_kernel<<<dim3(DD / 64, BS / 64), 128, 0, stream>>>(ctx, Wo, bo, q, Xb);

    layernorm_kernel<<<dim3(BS), 256, 0, stream>>>(Xb, gamma, beta, (float*)d_out);
}
